// VariationalGCNEncoder_5377299055295
// MI455X (gfx1250) — matrix operand order repaired and checked by reading, not compiled, not run on hardware
//
#include <hip/hip_runtime.h>
#include <hip/hip_bf16.h>

#define IN_C  256
#define OUT_C 64
#define HID   128

typedef __attribute__((ext_vector_type(16))) __bf16          v16bf;
typedef __attribute__((ext_vector_type(16))) unsigned short  v16u;
typedef __attribute__((ext_vector_type(8)))  unsigned short  v8u16;
typedef __attribute__((ext_vector_type(8)))  float           v8f;

__device__ __forceinline__ unsigned short f2bf_rne(float f) {
  union { float f; unsigned int u; } x; x.f = f;
  unsigned int u = x.u;
  u += 0x7FFFu + ((u >> 16) & 1u);   // round-to-nearest-even to bf16
  return (unsigned short)(u >> 16);
}

// ---------------- degree / norm ----------------
__global__ void init_deg(float* deg, int n) {
  int i = blockIdx.x * blockDim.x + threadIdx.x;
  if (i < n) deg[i] = 1.0f;                       // self-loop contributes 1
}

__global__ void count_deg(const int* __restrict__ dst, float* deg, int e) {
  int i = blockIdx.x * blockDim.x + threadIdx.x;
  if (i < e) atomicAdd(&deg[dst[i]], 1.0f);
}

__global__ void to_dinv(float* deg, int n) {
  int i = blockIdx.x * blockDim.x + threadIdx.x;
  if (i < n) deg[i] = rsqrtf(deg[i]);             // deg >= 1 always
}

__global__ void compute_norm(const int* __restrict__ src, const int* __restrict__ dst,
                             const float* __restrict__ dinv, float* __restrict__ nrm, int e) {
  int i = blockIdx.x * blockDim.x + threadIdx.x;
  if (i < e) nrm[i] = dinv[src[i]] * dinv[dst[i]];
}

// ---------------- bf16 packing ----------------
// row-major f32 -> row-major bf16 (for A matrices of the GEMMs)
__global__ void pack_bf16(const float* __restrict__ in, unsigned short* __restrict__ out,
                          long long total) {
  long long t = (long long)blockIdx.x * blockDim.x + threadIdx.x;
  if (t < total) out[t] = f2bf_rne(in[t]);
}

// W[K,NC] f32 row-major -> Wt[NC,K] bf16 (transposed: column contiguous in K)
__global__ void pack_bf16_T(const float* __restrict__ W, unsigned short* __restrict__ Wt,
                            int K, int NC) {
  int idx = blockIdx.x * blockDim.x + threadIdx.x;
  if (idx >= K * NC) return;
  int k = idx / NC, n = idx % NC;
  Wt[(size_t)n * K + k] = f2bf_rne(W[idx]);
}

// ---------- WMMA GEMM: C[M,NC] = A[M,K] @ B[K,NC] (+bias, +relu) -----------
// A pre-packed bf16 row-major, B pre-packed bf16 transposed [NC,K].
// Per ISA 7.12.2, each lane's 16-bf16 A fragment is two runs of 8 contiguous
// K values -> two b128 loads; same for the transposed B. No per-step VALU.
// block = (32, NC/16): one wave per 16x16 C tile; grid.x = M/16.
__global__ void gemm_bf16pk_wmma(const unsigned short* __restrict__ Apk,
                                 const unsigned short* __restrict__ Bt,
                                 const float* __restrict__ bias,
                                 float* __restrict__ C,
                                 int M, int K, int NC, int relu) {
  const int lane = threadIdx.x;       // 0..31
  const int half = lane >> 4;         // 0 or 1
  const int l15  = lane & 15;
  const int m0   = blockIdx.x * 16;
  const int n0   = threadIdx.y * 16;

  v8f acc = {};
  // A: lane row M=l15; elems 0..7 -> K = k0+8*half+(0..7); 8..15 -> +16
  const unsigned short* Arow = Apk + (size_t)(m0 + l15) * K + 8 * half;
  // B: lane col N=l15; elems 0..7 -> K = k0+8*half+(0..7); 8..15 -> +16
  const unsigned short* Brow = Bt + (size_t)(n0 + l15) * K + 8 * half;

  for (int k0 = 0; k0 < K; k0 += 32) {
    v8u16 alo = *(const v8u16*)(Arow + k0);
    v8u16 ahi = *(const v8u16*)(Arow + k0 + 16);
    v8u16 blo = *(const v8u16*)(Brow + k0);
    v8u16 bhi = *(const v8u16*)(Brow + k0 + 16);
    v16u av = __builtin_shufflevector(alo, ahi, 0,1,2,3,4,5,6,7,8,9,10,11,12,13,14,15);
    v16u bv = __builtin_shufflevector(blo, bhi, 0,1,2,3,4,5,6,7,8,9,10,11,12,13,14,15);
    acc = __builtin_amdgcn_wmma_f32_16x16x32_bf16(
        /*neg_a=*/false, __builtin_bit_cast(v16bf, av),
        /*neg_b=*/false, __builtin_bit_cast(v16bf, bv),
        /*c_mod=*/(short)0, acc, /*reuse_a=*/false, /*reuse_b=*/false);
  }

  const float bb = bias ? bias[n0 + l15] : 0.0f;
  // D layout: VGPR r holds M = r + 8*half, N = l15
  #pragma unroll
  for (int r = 0; r < 8; ++r) {
    int row = m0 + r + 8 * half;
    float v = acc[r] + bb;
    if (relu) v = fmaxf(v, 0.0f);
    C[(size_t)row * NC + n0 + l15] = v;
  }
}

// ---------------- aggregation ----------------
// Self-loop (non-atomic): out[i,:] = dinv[i]^2 * in[i,:]
__global__ void self_loop_init(const float* __restrict__ dinv,
                               const float* __restrict__ in,
                               float* __restrict__ out,
                               long long total, int c) {
  long long t = (long long)blockIdx.x * blockDim.x + threadIdx.x;
  if (t >= total) return;
  int i = (int)(t / c);
  float di = dinv[i];
  out[t] = di * di * in[t];
}

// Edge scatter-add: one wave per edge (all 32 lanes share one edge, so the
// edge scalars are wave-uniform -> readfirstlane). float4 gather + 4 coalesced
// f32 atomics; both source and destination rows are L2-resident (51 MB each).
__global__ void edge_agg128(const int* __restrict__ src,
                            const int* __restrict__ dst,
                            const float* __restrict__ nrm,
                            const float* __restrict__ in,
                            float* __restrict__ out, int e) {
  long long tid = (long long)blockIdx.x * blockDim.x + threadIdx.x;
  long long edge = tid >> 5;               // 128 floats / 4 = 32 lanes per edge
  if (edge >= e) return;
  int c = ((int)tid & 31) * 4;
  int s = __builtin_amdgcn_readfirstlane(src[edge]);
  int d = __builtin_amdgcn_readfirstlane(dst[edge]);
  float w = nrm[edge];
  const float* ip = in + (size_t)s * HID + c;
  __builtin_prefetch(ip, 0, 1);            // global_prefetch_b8 on gather row
  const float4 v = *(const float4*)ip;
  float* o = out + (size_t)d * HID + c;
  atomicAdd(o + 0, w * v.x);
  atomicAdd(o + 1, w * v.y);
  atomicAdd(o + 2, w * v.z);
  atomicAdd(o + 3, w * v.w);
}

__global__ void bias_relu(float* __restrict__ h, const float* __restrict__ b,
                          long long total, int c) {
  long long t = (long long)blockIdx.x * blockDim.x + threadIdx.x;
  if (t >= total) return;
  h[t] = fmaxf(h[t] + b[(int)(t % c)], 0.0f);
}

extern "C" void kernel_launch(void* const* d_in, const int* in_sizes, int n_in,
                              void* d_out, int out_size, void* d_ws, size_t ws_size,
                              hipStream_t stream) {
  const float* x   = (const float*)d_in[0];
  const int*   ei  = (const int*)d_in[1];
  const float* W1  = (const float*)d_in[2];
  const float* b1  = (const float*)d_in[3];
  const float* Wmu = (const float*)d_in[4];
  const float* bmu = (const float*)d_in[5];
  const float* Wls = (const float*)d_in[6];
  const float* bls = (const float*)d_in[7];

  const int N = in_sizes[0] / IN_C;     // 100000
  const int E = in_sizes[1] / 2;        // 1600000
  const int* src = ei;
  const int* dst = ei + E;

  float* out_mu = (float*)d_out;
  float* out_ls = out_mu + (size_t)N * OUT_C;

  // workspace layout
  float* dinv = (float*)d_ws;                                   // N f32
  size_t padN = ((size_t)N + 511) & ~(size_t)511;
  float* nrm  = dinv + padN;                                    // E f32
  size_t padE = ((size_t)E + 511) & ~(size_t)511;
  float* t    = nrm + padE;                                     // N*HID f32
  float* h    = t + (size_t)N * HID;                            // N*HID f32
  unsigned short* pk   = (unsigned short*)(h + (size_t)N * HID);// N*IN_C bf16 (x, then a)
  unsigned short* w1t  = pk + (size_t)N * IN_C;                 // IN_C*HID bf16
  unsigned short* wmut = w1t + IN_C * HID;                      // HID*OUT_C bf16
  unsigned short* wlst = wmut + HID * OUT_C;                    // HID*OUT_C bf16

  const long long NH  = (long long)N * HID;
  const long long NX  = (long long)N * IN_C;
  const int B = 256;
  const int gN    = (N + B - 1) / B;
  const int gE    = (E + B - 1) / B;
  const int gNH   = (int)((NH + B - 1) / B);
  const int gNX   = (int)((NX + B - 1) / B);
  const int gEdge = (int)(((long long)E * 32 + B - 1) / B);

  // GCN normalization + per-edge weights
  init_deg<<<gN, B, 0, stream>>>(dinv, N);
  count_deg<<<gE, B, 0, stream>>>(dst, dinv, E);
  to_dinv<<<gN, B, 0, stream>>>(dinv, N);
  compute_norm<<<gE, B, 0, stream>>>(src, dst, dinv, nrm, E);

  // one-shot bf16 packing (x row-major; weights transposed)
  pack_bf16<<<gNX, B, 0, stream>>>(x, pk, NX);
  pack_bf16_T<<<(IN_C * HID + B - 1) / B, B, 0, stream>>>(W1, w1t, IN_C, HID);
  pack_bf16_T<<<(HID * OUT_C + B - 1) / B, B, 0, stream>>>(Wmu, wmut, HID, OUT_C);
  pack_bf16_T<<<(HID * OUT_C + B - 1) / B, B, 0, stream>>>(Wls, wlst, HID, OUT_C);

  // Layer 1: t = x @ W1 (WMMA), aggregate, bias+relu
  gemm_bf16pk_wmma<<<dim3(N / 16), dim3(32, HID / 16), 0, stream>>>(
      pk, w1t, nullptr, t, N, IN_C, HID, 0);
  self_loop_init<<<gNH, B, 0, stream>>>(dinv, t, h, NH, HID);
  edge_agg128<<<gEdge, B, 0, stream>>>(src, dst, nrm, t, h, E);
  bias_relu<<<gNH, B, 0, stream>>>(h, b1, NH, HID);

  // Layer 2: aggregate ONCE (Agg(h@W) == Agg(h)@W), then two tiny WMMA GEMMs
  self_loop_init<<<gNH, B, 0, stream>>>(dinv, h, t, NH, HID);    // a := t
  edge_agg128<<<gEdge, B, 0, stream>>>(src, dst, nrm, h, t, E);

  pack_bf16<<<gNH, B, 0, stream>>>(t, pk, NH);                   // a -> bf16
  gemm_bf16pk_wmma<<<dim3(N / 16), dim3(32, OUT_C / 16), 0, stream>>>(
      pk, wmut, bmu, out_mu, N, HID, OUT_C, 0);
  gemm_bf16pk_wmma<<<dim3(N / 16), dim3(32, OUT_C / 16), 0, stream>>>(
      pk, wlst, bls, out_ls, N, HID, OUT_C, 0);
}
